// Attention_60473139528253
// MI455X (gfx1250) — compile-verified, matrix operand order checked by reading, not run on hardware
//
#include <hip/hip_runtime.h>
#include <hip/hip_bf16.h>

// ---------------------------------------------------------------------------
// Problem constants (from reference): B=2, S=1024, H=4096, NH=32, NKV=8, HD=128
// ---------------------------------------------------------------------------
#define B_   2
#define S_   1024
#define H_   4096
#define NH_  32
#define NKV_ 8
#define HD_  128
#define GROUP_ (NH_ / NKV_)

#define AS1 __attribute__((address_space(1)))
#define AS3 __attribute__((address_space(3)))

typedef __attribute__((ext_vector_type(16))) __bf16 v16bf;
typedef __attribute__((ext_vector_type(8)))  float  v8f;
typedef int v4i __attribute__((vector_size(16)));

union BFrag {
    v16bf v;
    uint4 u[2];
};

// Load one 16-bit A/B WMMA fragment (16 halves per lane) from a row pointer
// already offset by kbase.  Per CDNA5 ISA layout, halves j=0..7 hold
// K=kbase..kbase+7 and halves j=8..15 hold K=kbase+16..kbase+23.
__device__ __forceinline__ v16bf load_frag16(const __bf16* p) {
    BFrag f;
    f.u[0] = *(const uint4*)(p);
    f.u[1] = *(const uint4*)(p + 16);
    return f.v;
}

// --------------------------- CDNA5 async-to-LDS -----------------------------
// 16-byte global -> LDS async copy (ASYNCcnt-tracked) with safe fallback.
__device__ __forceinline__ void async_ld_b128(const __bf16* g, __bf16* l) {
#if defined(__AMDGCN__) && __has_builtin(__builtin_amdgcn_global_load_async_to_lds_b128)
    __builtin_amdgcn_global_load_async_to_lds_b128(
        (AS1 v4i*)g, (AS3 v4i*)l, 0, 0);
#else
    *(uint4*)l = *(const uint4*)g;
#endif
}

__device__ __forceinline__ void wait_async0() {
#if defined(__AMDGCN__) && __has_builtin(__builtin_amdgcn_s_wait_asynccnt)
    __builtin_amdgcn_s_wait_asynccnt(0);
#elif defined(__AMDGCN__)
    asm volatile("s_wait_asynccnt 0x0" ::: "memory");
#endif
}

// ---------------------------------------------------------------------------
// f32 -> bf16 conversion
// ---------------------------------------------------------------------------
__global__ void cvt_f32_bf16(const float* __restrict__ src,
                             __bf16* __restrict__ dst, int n) {
    int i = blockIdx.x * blockDim.x + threadIdx.x;
    if (i < n) dst[i] = (__bf16)src[i];
}

// ---------------------------------------------------------------------------
// Generic GEMM:  C[M,N] = A[M,K] * W[N,K]^T   (bf16 inputs, f32 out)
// Block = 256 threads (8 waves).  Wave computes a 16(M) x 64(N) tile.
// K-loop unrolled x2 with ping-pong fragment sets: loads for the next K-step
// land directly in the alternate register set (no v_mov rotation), and WMMAs
// overlap outstanding loads via partial loadcnt waits.
// Requires K a multiple of 64 (true for all uses here: K = 4096).
// grid.x = M/16, grid.y = N/(64*8)
// ---------------------------------------------------------------------------
__global__ __launch_bounds__(256)
void gemm_bf16_wmma(const __bf16* __restrict__ A, const __bf16* __restrict__ W,
                    float* __restrict__ C, int M, int N, int K) {
    const int lane  = threadIdx.x & 31;
    const int wave  = threadIdx.x >> 5;
    const int m0    = blockIdx.x * 16;
    const int n0    = (blockIdx.y * 8 + wave) * 64;
    if (m0 >= M || n0 >= N) return;

    const int mrow  = lane & 15;
    const int kbase = (lane < 16) ? 0 : 8;

    v8f acc[4] = {};

    const __bf16* arow = A + (size_t)(m0 + mrow) * K + kbase;
    const __bf16* wrow = W + (size_t)(n0 + mrow) * K + kbase;

    v16bf af0, af1, b0[4], b1[4];

    // prologue: fragments for k = 0
    af0 = load_frag16(arow);
#pragma unroll
    for (int t = 0; t < 4; ++t) b0[t] = load_frag16(wrow + (size_t)t * 16 * K);

    int k = 0;
    for (; k + 64 < K; k += 64) {
        // issue loads for k+32 into the alternate set
        af1 = load_frag16(arow + k + 32);
#pragma unroll
        for (int t = 0; t < 4; ++t)
            b1[t] = load_frag16(wrow + (size_t)t * 16 * K + k + 32);
        __builtin_prefetch(arow + k + 256, 0, 1);  // global_prefetch_b8

        // math for k overlaps those loads
#pragma unroll
        for (int t = 0; t < 4; ++t)
            acc[t] = __builtin_amdgcn_wmma_f32_16x16x32_bf16(
                false, af0, false, b0[t], (short)0, acc[t], false, false);

        // issue loads for k+64 back into set 0
        af0 = load_frag16(arow + k + 64);
#pragma unroll
        for (int t = 0; t < 4; ++t)
            b0[t] = load_frag16(wrow + (size_t)t * 16 * K + k + 64);

        // math for k+32 overlaps those loads
#pragma unroll
        for (int t = 0; t < 4; ++t)
            acc[t] = __builtin_amdgcn_wmma_f32_16x16x32_bf16(
                false, af1, false, b1[t], (short)0, acc[t], false, false);
    }

    // peeled tail: k = K-64  (no out-of-bounds load issue)
    af1 = load_frag16(arow + k + 32);
#pragma unroll
    for (int t = 0; t < 4; ++t)
        b1[t] = load_frag16(wrow + (size_t)t * 16 * K + k + 32);
#pragma unroll
    for (int t = 0; t < 4; ++t)
        acc[t] = __builtin_amdgcn_wmma_f32_16x16x32_bf16(
            false, af0, false, b0[t], (short)0, acc[t], false, false);
#pragma unroll
    for (int t = 0; t < 4; ++t)
        acc[t] = __builtin_amdgcn_wmma_f32_16x16x32_bf16(
            false, af1, false, b1[t], (short)0, acc[t], false, false);

    const int ncol = lane & 15;
    const int mhi  = (lane < 16) ? 0 : 8;
#pragma unroll
    for (int t = 0; t < 4; ++t)
#pragma unroll
        for (int r = 0; r < 8; ++r)
            C[(size_t)(m0 + r + mhi) * N + n0 + t * 16 + ncol] = acc[t][r];
}

// ---------------------------------------------------------------------------
// RoPE on Q:  qf32 [B,S,NH,HD] -> qbf [B,NH,S,HD] (bf16, rotated)
// ---------------------------------------------------------------------------
__global__ void rope_q_kernel(const float* __restrict__ qf,
                              const float* __restrict__ cosb,
                              const float* __restrict__ sinb,
                              __bf16* __restrict__ qbf) {
    int i = blockIdx.x * blockDim.x + threadIdx.x;
    const int total = B_ * S_ * NH_ * (HD_ / 2);
    if (i >= total) return;
    int d = i & 63; int t = i >> 6;
    int h = t % NH_; t /= NH_;
    int s = t % S_;  int b = t / S_;

    float q0 = qf[(((size_t)b * S_ + s) * NH_ + h) * HD_ + d];
    float q1 = qf[(((size_t)b * S_ + s) * NH_ + h) * HD_ + d + 64];
    float c0 = cosb[(size_t)s * HD_ + d],      c1 = cosb[(size_t)s * HD_ + d + 64];
    float s0 = sinb[(size_t)s * HD_ + d],      s1 = sinb[(size_t)s * HD_ + d + 64];
    float o0 = q0 * c0 - q1 * s0;
    float o1 = q1 * c1 + q0 * s1;
    size_t ob = (((size_t)b * NH_ + h) * S_ + s) * HD_;
    qbf[ob + d]      = (__bf16)o0;
    qbf[ob + d + 64] = (__bf16)o1;
}

// RoPE on K:  kf32 [B,S,NKV,HD] -> k_ret f32 [B,NKV,S,HD]  +  kbf bf16
__global__ void rope_k_kernel(const float* __restrict__ kf,
                              const float* __restrict__ cosb,
                              const float* __restrict__ sinb,
                              float* __restrict__ kret,
                              __bf16* __restrict__ kbf) {
    int i = blockIdx.x * blockDim.x + threadIdx.x;
    const int total = B_ * S_ * NKV_ * (HD_ / 2);
    if (i >= total) return;
    int d = i & 63; int t = i >> 6;
    int kv = t % NKV_; t /= NKV_;
    int s = t % S_;    int b = t / S_;

    float k0 = kf[(((size_t)b * S_ + s) * NKV_ + kv) * HD_ + d];
    float k1 = kf[(((size_t)b * S_ + s) * NKV_ + kv) * HD_ + d + 64];
    float c0 = cosb[(size_t)s * HD_ + d],      c1 = cosb[(size_t)s * HD_ + d + 64];
    float s0 = sinb[(size_t)s * HD_ + d],      s1 = sinb[(size_t)s * HD_ + d + 64];
    float o0 = k0 * c0 - k1 * s0;
    float o1 = k1 * c1 + k0 * s1;
    size_t ob = (((size_t)b * NKV_ + kv) * S_ + s) * HD_;
    kret[ob + d]      = o0;
    kret[ob + d + 64] = o1;
    kbf[ob + d]       = (__bf16)o0;
    kbf[ob + d + 64]  = (__bf16)o1;
}

// V: transpose [B,S,NKV,HD] -> [B,NKV,S,HD] f32 (v_ret) and bf16 copy
__global__ void copy_v_kernel(const float* __restrict__ vf,
                              float* __restrict__ vret,
                              __bf16* __restrict__ vbf) {
    int i = blockIdx.x * blockDim.x + threadIdx.x;
    const int total = B_ * S_ * NKV_ * HD_;
    if (i >= total) return;
    int d = i % HD_; int t = i / HD_;
    int kv = t % NKV_; t /= NKV_;
    int s = t % S_;    int b = t / S_;
    float val = vf[(((size_t)b * S_ + s) * NKV_ + kv) * HD_ + d];
    size_t o = (((size_t)b * NKV_ + kv) * S_ + s) * HD_ + d;
    vret[o] = val;
    vbf[o]  = (__bf16)val;
}

// ---------------------------------------------------------------------------
// Flash attention (causal, GQA).
// Block = 128 threads (4 waves).  grid = (B*NH, S/64).
// Double-buffered LDS: async-to-LDS copies (ASYNCcnt) for tile kt+1 are
// issued before the math of tile kt, so the Tensor/async engine overlaps the
// WMMA work; s_wait_asynccnt only at the end of each iteration.
// ---------------------------------------------------------------------------
__global__ __launch_bounds__(128)
void flash_attn_wmma(const __bf16* __restrict__ qbf,
                     const __bf16* __restrict__ kbf,
                     const __bf16* __restrict__ vbf,
                     __bf16* __restrict__ ctxbf) {
    __shared__ __bf16 Kt[2][32][128];    // [buf][key][d]
    __shared__ __bf16 Vraw[2][32][128];  // [buf][key][d] async landing buffer
    __shared__ __bf16 Vt[2][128][40];    // [buf][d][key], padded stride
    __shared__ __bf16 Pb[4][16][32];     // per-wave P scratch [row][key]

    const int lane  = threadIdx.x & 31;
    const int wave  = threadIdx.x >> 5;
    const int bh    = blockIdx.x;
    const int b     = bh / NH_;
    const int h     = bh % NH_;
    const int kv    = h / GROUP_;
    const int qtile = blockIdx.y;
    const int qb    = qtile * 64 + wave * 16;   // first q row of this wave

    const int mrow  = lane & 15;
    const int kbase = (lane < 16) ? 0 : 8;
    const int mhi   = (lane < 16) ? 0 : 8;

    const __bf16* kh = kbf + ((size_t)b * NKV_ + kv) * S_ * HD_;
    const __bf16* vh = vbf + ((size_t)b * NKV_ + kv) * S_ * HD_;

    // Q fragments: 16 rows x 128 dims = 4 A-frags, held in registers.
    const __bf16* qptr =
        qbf + (((size_t)b * NH_ + h) * S_ + (qb + mrow)) * HD_ + kbase;
    v16bf qf[4];
#pragma unroll
    for (int f = 0; f < 4; ++f) qf[f] = load_frag16(qptr + f * 32);

    float mrun[8], lrun[8];
#pragma unroll
    for (int r = 0; r < 8; ++r) { mrun[r] = -3.0e38f; lrun[r] = 0.0f; }
    v8f accp[8] = {};

    const float scale = 0.08838834764831845f;  // 1/sqrt(128)
    const int nkt = 2 * qtile + 2;             // key tiles covering causal range

    // async stage of one 32-key K/V tile into LDS buffer `buf`
    auto stage = [&](int kt, int buf) {
        const int k0 = kt * 32;
        for (int idx = threadIdx.x; idx < 32 * 16; idx += 128) {
            const int row = idx >> 4, c = (idx & 15) * 8;
            async_ld_b128(kh + (size_t)(k0 + row) * HD_ + c, &Kt[buf][row][c]);
            async_ld_b128(vh + (size_t)(k0 + row) * HD_ + c, &Vraw[buf][row][c]);
        }
    };

    // prologue: stage tile 0
    stage(0, 0);
    wait_async0();
    __syncthreads();

    for (int kt = 0; kt < nkt; ++kt) {
        const int cur = kt & 1;
        const int k0  = kt * 32;

        // issue async loads for the NEXT tile into the alternate buffer;
        // they run concurrently with all the math below.
        if (kt + 1 < nkt) stage(kt + 1, cur ^ 1);

        // Transpose V inside LDS: Vraw[cur][32][128] -> Vt[cur][128][40]
        for (int idx = threadIdx.x; idx < 32 * 16; idx += 128) {
            const int row = idx >> 4, c = (idx & 15) * 8;
            uint4 vd = *(const uint4*)&Vraw[cur][row][c];
            const __bf16* ve = (const __bf16*)&vd;
#pragma unroll
            for (int e = 0; e < 8; ++e) Vt[cur][c + e][row] = ve[e];
        }
        __syncthreads();

        // scores: 16 rows x 32 keys = two 16x16 f32 tiles, K-dim = HD (4 steps)
        v8f sc[2] = {};
#pragma unroll
        for (int half = 0; half < 2; ++half) {
            const int keyl = half * 16 + (lane & 15);
#pragma unroll
            for (int f = 0; f < 4; ++f) {
                BFrag kfr;
                kfr.u[0] = *(const uint4*)&Kt[cur][keyl][f * 32 + kbase];
                kfr.u[1] = *(const uint4*)&Kt[cur][keyl][f * 32 + kbase + 16];
                sc[half] = __builtin_amdgcn_wmma_f32_16x16x32_bf16(
                    false, qf[f], false, kfr.v, (short)0, sc[half], false, false);
            }
        }

        // online softmax with causal mask
#pragma unroll
        for (int r = 0; r < 8; ++r) {
            const int rowg = qb + r + mhi;
            float s0 = sc[0][r] * scale;
            float s1 = sc[1][r] * scale;
            if (k0 + (lane & 15) > rowg)      s0 = -3.0e38f;
            if (k0 + 16 + (lane & 15) > rowg) s1 = -3.0e38f;

            float tmax = fmaxf(s0, s1);
            tmax = fmaxf(tmax, __shfl_xor(tmax, 1));
            tmax = fmaxf(tmax, __shfl_xor(tmax, 2));
            tmax = fmaxf(tmax, __shfl_xor(tmax, 4));
            tmax = fmaxf(tmax, __shfl_xor(tmax, 8));

            const float Mn = fmaxf(mrun[r], tmax);
            const float sf = __expf(mrun[r] - Mn);
            mrun[r] = Mn;
            lrun[r] *= sf;
#pragma unroll
            for (int t = 0; t < 8; ++t) accp[t][r] *= sf;

            const float p0 = __expf(s0 - Mn);
            const float p1 = __expf(s1 - Mn);
            float ts = p0 + p1;
            ts += __shfl_xor(ts, 1);
            ts += __shfl_xor(ts, 2);
            ts += __shfl_xor(ts, 4);
            ts += __shfl_xor(ts, 8);
            lrun[r] += ts;

            const int col = lane & 15;
            Pb[wave][r + mhi][col]      = (__bf16)p0;
            Pb[wave][r + mhi][16 + col] = (__bf16)p1;
        }

        // re-layout P (per-wave LDS round trip) into an A-fragment
        BFrag pf;
        pf.u[0] = *(const uint4*)&Pb[wave][mrow][kbase];
        pf.u[1] = *(const uint4*)&Pb[wave][mrow][kbase + 16];

        // ctx += P(16x32) x V(32x128): 8 WMMAs over 16-wide dim tiles
#pragma unroll
        for (int t = 0; t < 8; ++t) {
            const int d = t * 16 + (lane & 15);
            BFrag vfr;
            vfr.u[0] = *(const uint4*)&Vt[cur][d][kbase];
            vfr.u[1] = *(const uint4*)&Vt[cur][d][kbase + 16];
            accp[t] = __builtin_amdgcn_wmma_f32_16x16x32_bf16(
                false, pf.v, false, vfr.v, (short)0, accp[t], false, false);
        }

        // next tile's async copies must have landed before the next iteration
        wait_async0();
        __syncthreads();
    }

    // epilogue: normalize and emit ctx bf16 in [B, S, NH*HD] layout
#pragma unroll
    for (int t = 0; t < 8; ++t)
#pragma unroll
        for (int r = 0; r < 8; ++r) {
            const int srow = qb + r + mhi;
            const float val = accp[t][r] / lrun[r];
            ctxbf[((size_t)b * S_ + srow) * (NH_ * HD_) + h * HD_ + t * 16 +
                  (lane & 15)] = (__bf16)val;
        }
}

// ---------------------------------------------------------------------------
// Launch
// ---------------------------------------------------------------------------
extern "C" void kernel_launch(void* const* d_in, const int* in_sizes, int n_in,
                              void* d_out, int out_size, void* d_ws,
                              size_t ws_size, hipStream_t stream) {
    (void)in_sizes; (void)n_in; (void)out_size; (void)ws_size;

    const float* hidden = (const float*)d_in[0];
    const float* cosb   = (const float*)d_in[1];
    const float* sinb   = (const float*)d_in[2];
    const float* wq     = (const float*)d_in[3];
    const float* wk     = (const float*)d_in[4];
    const float* wv     = (const float*)d_in[5];
    const float* wd     = (const float*)d_in[6];

    float* out_f32 = (float*)d_out;                                   // [B,S,H]
    float* k_ret   = out_f32 + (size_t)B_ * S_ * H_;                  // [B,NKV,S,HD]
    float* v_ret   = k_ret + (size_t)B_ * NKV_ * S_ * HD_;            // [B,NKV,S,HD]

    // workspace arena
    char* w = (char*)d_ws;
    size_t off = 0;
    auto arena = [&](size_t bytes) -> void* {
        void* p = w + off;
        off = (off + bytes + 255) & ~(size_t)255;
        return p;
    };
    const size_t NBS = (size_t)B_ * S_;               // 2048 rows
    __bf16* hbf   = (__bf16*)arena(NBS * H_ * 2);
    __bf16* wqbf  = (__bf16*)arena((size_t)NH_ * HD_ * H_ * 2);
    __bf16* wkbf  = (__bf16*)arena((size_t)NKV_ * HD_ * H_ * 2);
    __bf16* wvbf  = (__bf16*)arena((size_t)NKV_ * HD_ * H_ * 2);
    __bf16* wdbf  = (__bf16*)arena((size_t)H_ * H_ * 2);
    float*  qf32  = (float*)arena(NBS * NH_ * HD_ * 4);
    float*  kf32  = (float*)arena(NBS * NKV_ * HD_ * 4);
    float*  vf32  = (float*)arena(NBS * NKV_ * HD_ * 4);
    __bf16* qbf   = (__bf16*)arena(NBS * NH_ * HD_ * 2);
    __bf16* kbf   = (__bf16*)arena(NBS * NKV_ * HD_ * 2);
    __bf16* vbf   = (__bf16*)arena(NBS * NKV_ * HD_ * 2);
    __bf16* ctxbf = (__bf16*)arena(NBS * NH_ * HD_ * 2);

    // 1) downconvert inputs to bf16
    auto cvt = [&](const float* s, __bf16* d, int n) {
        cvt_f32_bf16<<<(n + 255) / 256, 256, 0, stream>>>(s, d, n);
    };
    cvt(hidden, hbf,  (int)(NBS * H_));
    cvt(wq,     wqbf, NH_ * HD_ * H_);
    cvt(wk,     wkbf, NKV_ * HD_ * H_);
    cvt(wv,     wvbf, NKV_ * HD_ * H_);
    cvt(wd,     wdbf, H_ * H_);

    // 2) QKV projections  (M=2048, K=4096)
    gemm_bf16_wmma<<<dim3(128, 8), 256, 0, stream>>>(hbf, wqbf, qf32,
                                                     (int)NBS, NH_ * HD_, H_);
    gemm_bf16_wmma<<<dim3(128, 2), 256, 0, stream>>>(hbf, wkbf, kf32,
                                                     (int)NBS, NKV_ * HD_, H_);
    gemm_bf16_wmma<<<dim3(128, 2), 256, 0, stream>>>(hbf, wvbf, vf32,
                                                     (int)NBS, NKV_ * HD_, H_);

    // 3) RoPE + layout transforms + kv-cache outputs
    {
        int n = B_ * S_ * NH_ * (HD_ / 2);
        rope_q_kernel<<<(n + 255) / 256, 256, 0, stream>>>(qf32, cosb, sinb, qbf);
    }
    {
        int n = B_ * S_ * NKV_ * (HD_ / 2);
        rope_k_kernel<<<(n + 255) / 256, 256, 0, stream>>>(kf32, cosb, sinb,
                                                           k_ret, kbf);
    }
    {
        int n = B_ * S_ * NKV_ * HD_;
        copy_v_kernel<<<(n + 255) / 256, 256, 0, stream>>>(vf32, v_ret, vbf);
    }

    // 4) causal GQA flash attention -> ctx bf16
    flash_attn_wmma<<<dim3(B_ * NH_, S_ / 64), 128, 0, stream>>>(qbf, kbf, vbf,
                                                                 ctxbf);

    // 5) output projection straight into d_out (f32)
    gemm_bf16_wmma<<<dim3(128, 8), 256, 0, stream>>>(ctxbf, wdbf, out_f32,
                                                     (int)NBS, H_, H_);
}
